// VectorQuantizer_17360257810582
// MI455X (gfx1250) — matrix operand order repaired and checked by reading, not compiled, not run on hardware
//
#include <hip/hip_runtime.h>
#include <hip/hip_bf16.h>

typedef __attribute__((ext_vector_type(16))) __bf16 v16bf;
typedef __attribute__((ext_vector_type(8)))  float  v8f;

#define D      512
#define NROWS  8192
#define KCODES 8192
#define MT     64            // rows of X per block
#define NT     64            // codebook columns per tile
#define XP     520           // padded X LDS row (elements) -> 1040 B, 16B aligned
#define EPAD   72            // padded E LDS row for 64-dim slice -> 144 B, 16B aligned
#define KSTEP  64            // dims per barrier step
#define NSTEP  (D / KSTEP)   // 8 steps per tile

#define XBYTES (2 * MT * XP * 2)        // 133120 B : Xhi + Xlo (persistent)
#define ELOOFF (NT * EPAD * 2)          //   9216 B : lo region offset in a buffer
#define EBUF   (2 * NT * EPAD * 2)      //  18432 B : one (Ehi+Elo) slice buffer
#define SMEMSZ (XBYTES + 2 * EBUF)      // 169984 B total dynamic LDS

union Frag  { v16bf v; uint4 u[2]; };
union Pack4 { __bf16 b[4]; uint2 u; };

// ---- CDNA5 async global->LDS copy (ASYNCcnt-tracked, bypasses VGPRs) ----
__device__ __forceinline__ unsigned lds_off(const void* p) {
    return (unsigned)(reinterpret_cast<uintptr_t>(p)); // low 32 bits = LDS byte offset
}
__device__ __forceinline__ void async_ld16(unsigned lds_addr, const __bf16* g) {
    asm volatile("global_load_async_to_lds_b128 %0, %1, off"
                 :: "v"(lds_addr), "v"(g) : "memory");
}
__device__ __forceinline__ void wait_async0() {
    asm volatile("s_wait_asynccnt 0x0" ::: "memory");
}

// ---------------------------------------------------------------------------
// Kernel 1: norms for X rows (0..8191) and E rows (8192..16383); for E rows
// also emit bf16 (hi,lo) split arrays so the hot loop has zero converts.
// ---------------------------------------------------------------------------
__global__ void vq_prep(const float* __restrict__ X, const float* __restrict__ E,
                        float* __restrict__ norms,
                        __bf16* __restrict__ Ehi, __bf16* __restrict__ Elo) {
    int row = blockIdx.x;                       // 0..16383
    int t   = threadIdx.x;                      // 128 threads x float4 = 512
    float4 a;
    if (row < NROWS) {
        a = ((const float4*)(X + (size_t)row * D))[t];
    } else {
        int r = row - NROWS;
        a = ((const float4*)(E + (size_t)r * D))[t];
        float v[4] = {a.x, a.y, a.z, a.w};
        Pack4 ph, pl;
#pragma unroll
        for (int q = 0; q < 4; q++) {
            ph.b[q] = (__bf16)v[q];
            pl.b[q] = (__bf16)(v[q] - (float)ph.b[q]);
        }
        *(uint2*)&Ehi[(size_t)r * D + t * 4] = ph.u;
        *(uint2*)&Elo[(size_t)r * D + t * 4] = pl.u;
    }
    float s = a.x*a.x + a.y*a.y + a.z*a.z + a.w*a.w;
    __shared__ float red[128];
    red[t] = s; __syncthreads();
    for (int off = 64; off > 0; off >>= 1) {
        if (t < off) red[t] += red[t + off];
        __syncthreads();
    }
    if (t == 0) norms[row] = red[0];
}

// ---------------------------------------------------------------------------
// Kernel 2: split-bf16 WMMA GEMM S = X·E^T fused with per-row argmin.
// 64-dim double-buffered E slices streamed L2->LDS via async copies;
// 12 WMMAs per barrier. X tile (hi/lo) persistent in LDS.
// ---------------------------------------------------------------------------
__global__ __launch_bounds__(256)
void vq_main(const float* __restrict__ X,
             const __bf16* __restrict__ Ehi, const __bf16* __restrict__ Elo,
             const float* __restrict__ norms,
             int*   __restrict__ bestIdxOut,
             float* __restrict__ outIdxF,
             float* __restrict__ outMinD) {
    extern __shared__ char smem[];
    __bf16* Xh = (__bf16*)smem;                  // MT*XP
    __bf16* Xl = Xh + MT * XP;                   // MT*XP

    const int tid  = threadIdx.x;
    const int lane = tid & 31;
    const int wave = tid >> 5;
    const int mw   = wave & 3;
    const int nw   = wave >> 2;
    const int m0   = blockIdx.x * MT;

    // async staging geometry: 4 threads per codebook row, 16 elems (32 B) each
    const int er = tid >> 2;                     // 0..63 tile row
    const int ec = (tid & 3) * 16;               // element offset in 64-dim slice
    const unsigned ebase = lds_off(smem) + XBYTES + (unsigned)(er * EPAD + ec) * 2;
    const __bf16* gH = Ehi + (size_t)er * D + ec;
    const __bf16* gL = Elo + (size_t)er * D + ec;

    // prologue: issue first slice (n0=0,d0=0) into buffer 0 (overlaps X staging)
    async_ld16(ebase,               gH);
    async_ld16(ebase + 16,          gH + 8);
    async_ld16(ebase + ELOOFF,      gL);
    async_ld16(ebase + ELOOFF + 16, gL + 8);

    // ---- stage X tile once: fp32 -> (hi, lo) bf16 split in LDS ----
    for (int i = tid; i < MT * (D / 4); i += 256) {
        int r  = i >> 7;
        int d4 = i & 127;
        float4 x = ((const float4*)(X + (size_t)(m0 + r) * D))[d4];
        float vals[4] = {x.x, x.y, x.z, x.w};
        int base = r * XP + d4 * 4;
#pragma unroll
        for (int q = 0; q < 4; q++) {
            __bf16 h = (__bf16)vals[q];
            Xh[base + q] = h;
            Xl[base + q] = (__bf16)(vals[q] - (float)h);
        }
    }
    __syncthreads();

    float bestd[8];
    int   bestk[8];
#pragma unroll
    for (int j = 0; j < 8; j++) { bestd[j] = 3.4e38f; bestk[j] = 0; }

    const int arow  = mw * 16 + (lane & 15);
    const int asel  = (lane >> 4) * 16;
    const int bcol0 = nw * 32 + (lane & 15);
    const int bcol1 = bcol0 + 16;
    const int bksel = (lane >> 4) * 16;

    for (int n0 = 0; n0 < KCODES; n0 += NT) {
        v8f c0 = {}; v8f c1 = {};
#pragma unroll 1
        for (int i = 0; i < NSTEP; i++) {
            const int bb = i & 1;
            const int d0 = i * KSTEP;

            wait_async0();            // slice i landed (async loads complete in order)
            __syncthreads();          // visible to all; readers of other buffer done

            // prefetch next slice into the other buffer (overlaps WMMAs)
            if (!(n0 == KCODES - NT && i == NSTEP - 1)) {
                int nn0 = (i == NSTEP - 1) ? n0 + NT : n0;
                int nd0 = (i == NSTEP - 1) ? 0 : d0 + KSTEP;
                unsigned lb = ebase + (unsigned)((i + 1) & 1) * EBUF;
                size_t  go = (size_t)nn0 * D + nd0;
                async_ld16(lb,               gH + go);
                async_ld16(lb + 16,          gH + go + 8);
                async_ld16(lb + ELOOFF,      gL + go);
                async_ld16(lb + ELOOFF + 16, gL + go + 8);
            }

            const __bf16* Eh = (const __bf16*)(smem + XBYTES + bb * EBUF);
            const __bf16* El = Eh + NT * EPAD;

#pragma unroll
            for (int sub = 0; sub < 2; sub++) {
                const int ds = d0 + sub * 32;    // dims within X row
                const int es = sub * 32;         // dims within E slice row

                // A fragments (hi, lo)
                Frag ah, al;
                const __bf16* xph = &Xh[arow * XP + ds + asel];
                ah.u[0] = *(const uint4*)(xph);
                ah.u[1] = *(const uint4*)(xph + 8);
                const __bf16* xpl = &Xl[arow * XP + ds + asel];
                al.u[0] = *(const uint4*)(xpl);
                al.u[1] = *(const uint4*)(xpl + 8);

                // B fragments: contiguous codebook-row segments
                Frag b0h, b0l, b1h, b1l;
                {
                    const uint4* p = (const uint4*)&Eh[bcol0 * EPAD + es + bksel];
                    b0h.u[0] = p[0]; b0h.u[1] = p[1];
                    const uint4* q = (const uint4*)&El[bcol0 * EPAD + es + bksel];
                    b0l.u[0] = q[0]; b0l.u[1] = q[1];
                    const uint4* r = (const uint4*)&Eh[bcol1 * EPAD + es + bksel];
                    b1h.u[0] = r[0]; b1h.u[1] = r[1];
                    const uint4* s = (const uint4*)&El[bcol1 * EPAD + es + bksel];
                    b1l.u[0] = s[0]; b1l.u[1] = s[1];
                }

                // 3-term split-bf16: hi*hi + hi*lo + lo*hi, f32 accumulate
                c0 = __builtin_amdgcn_wmma_f32_16x16x32_bf16(false, ah.v, false, b0h.v, (short)0, c0, false, false);
                c0 = __builtin_amdgcn_wmma_f32_16x16x32_bf16(false, ah.v, false, b0l.v, (short)0, c0, false, false);
                c0 = __builtin_amdgcn_wmma_f32_16x16x32_bf16(false, al.v, false, b0h.v, (short)0, c0, false, false);
                c1 = __builtin_amdgcn_wmma_f32_16x16x32_bf16(false, ah.v, false, b1h.v, (short)0, c1, false, false);
                c1 = __builtin_amdgcn_wmma_f32_16x16x32_bf16(false, ah.v, false, b1l.v, (short)0, c1, false, false);
                c1 = __builtin_amdgcn_wmma_f32_16x16x32_bf16(false, al.v, false, b1h.v, (short)0, c1, false, false);
            }
        }

        // distance + running argmin (k ascending -> strict < keeps first index)
        float e0 = norms[NROWS + n0 + bcol0];
        float e1 = norms[NROWS + n0 + bcol1];
        int k0 = n0 + bcol0, k1 = n0 + bcol1;
#pragma unroll
        for (int j = 0; j < 8; j++) {
            float dd0 = e0 - 2.0f * c0[j];
            if (dd0 < bestd[j]) { bestd[j] = dd0; bestk[j] = k0; }
            float dd1 = e1 - 2.0f * c1[j];
            if (dd1 < bestd[j]) { bestd[j] = dd1; bestk[j] = k1; }
        }
    }

    // ---- cross-lane/wave argmin reduction (reuse LDS: 64 rows x 32 slots) ----
    __syncthreads();
    float* redD = (float*)smem;
    int*   redI = (int*)(smem + MT * 32 * sizeof(float));
    {
        int rbase = mw * 16 + (lane >> 4) * 8;
        int slot  = nw * 16 + (lane & 15);
#pragma unroll
        for (int j = 0; j < 8; j++) {
            redD[(rbase + j) * 32 + slot] = bestd[j];
            redI[(rbase + j) * 32 + slot] = bestk[j];
        }
    }
    __syncthreads();
    if (tid < MT) {
        float bd = redD[tid * 32];
        int   bk = redI[tid * 32];
        for (int s = 1; s < 32; s++) {
            float d = redD[tid * 32 + s];
            int   k = redI[tid * 32 + s];
            if (d < bd || (d == bd && k < bk)) { bd = d; bk = k; }
        }
        int row = m0 + tid;
        bestIdxOut[row] = bk;
        outIdxF[row]    = (float)bk;
        outMinD[row]    = norms[row] + bd;       // + ||x||^2
    }
}

// ---------------------------------------------------------------------------
// Kernel 3: gather quantized rows, straight-through output, per-row loss partial
// ---------------------------------------------------------------------------
__global__ void vq_gather(const float* __restrict__ X, const float* __restrict__ E,
                          const int* __restrict__ idx, float* __restrict__ outQ,
                          float* __restrict__ partial) {
    int row = blockIdx.x;
    int t   = threadIdx.x;                      // 128 threads x float4 = 512
    int k   = idx[row];
    float4 ev = ((const float4*)(E + (size_t)k   * D))[t];
    float4 xv = ((const float4*)(X + (size_t)row * D))[t];
    float dx = ev.x - xv.x, dy = ev.y - xv.y, dz = ev.z - xv.z, dw = ev.w - xv.w;
    float4 q;                                   // x + (q - x), as reference STE
    q.x = xv.x + dx; q.y = xv.y + dy; q.z = xv.z + dz; q.w = xv.w + dw;
    ((float4*)(outQ + (size_t)row * D))[t] = q;
    float s = dx*dx + dy*dy + dz*dz + dw*dw;
    __shared__ float red[128];
    red[t] = s; __syncthreads();
    for (int off = 64; off > 0; off >>= 1) {
        if (t < off) red[t] += red[t + off];
        __syncthreads();
    }
    if (t == 0) partial[row] = red[0];
}

// ---------------------------------------------------------------------------
// Kernel 4: deterministic final loss reduction
// ---------------------------------------------------------------------------
__global__ void vq_finalize(const float* __restrict__ partial,
                            float* __restrict__ outLoss, float* __restrict__ outCommit) {
    int t = threadIdx.x;                        // 256
    double s = 0.0;
    for (int i = t; i < NROWS; i += 256) s += (double)partial[i];
    __shared__ double red[256];
    red[t] = s; __syncthreads();
    for (int off = 128; off > 0; off >>= 1) {
        if (t < off) red[t] += red[t + off];
        __syncthreads();
    }
    if (t == 0) {
        double S = red[0];
        *outCommit = (float)S;
        *outLoss   = (float)(1.25 * S);
    }
}

// ---------------------------------------------------------------------------
extern "C" void kernel_launch(void* const* d_in, const int* in_sizes, int n_in,
                              void* d_out, int out_size, void* d_ws, size_t ws_size,
                              hipStream_t stream) {
    (void)in_sizes; (void)n_in; (void)out_size; (void)ws_size;
    const float* X = (const float*)d_in[0];     // inputs  [8,1024,512] fp32
    const float* E = (const float*)d_in[1];     // codebook [8192,512]  fp32

    float* out       = (float*)d_out;
    float* outQ      = out;                             // [4194304]
    float* outLoss   = out + (size_t)NROWS * D;         // [1]
    float* outIdx    = outLoss + 1;                     // [8192]
    float* outMinD   = outIdx + NROWS;                  // [8192]
    float* outCommit = outMinD + NROWS;                 // [1]

    char*  wsb     = (char*)d_ws;
    float* norms   = (float*)wsb;                       // [16384] f32
    float* partial = (float*)(wsb + 65536);             // [8192]  f32
    int*   bestIdx = (int*)  (wsb + 98304);             // [8192]  i32
    __bf16* Ehi    = (__bf16*)(wsb + 131072);           // [8192*512] bf16 (8 MB)
    __bf16* Elo    = Ehi + (size_t)KCODES * D;          // [8192*512] bf16 (8 MB)

    (void)hipFuncSetAttribute((const void*)vq_main,
                              hipFuncAttributeMaxDynamicSharedMemorySize, (int)SMEMSZ);

    vq_prep    <<<2 * NROWS,  128, 0,      stream>>>(X, E, norms, Ehi, Elo);
    vq_main    <<<NROWS / MT, 256, SMEMSZ, stream>>>(X, Ehi, Elo, norms, bestIdx, outIdx, outMinD);
    vq_gather  <<<NROWS,      128, 0,      stream>>>(X, E, bestIdx, outQ, partial);
    vq_finalize<<<1,          256, 0,      stream>>>(partial, outLoss, outCommit);
}
